// PagedAttention_73924977099014
// MI455X (gfx1250) — matrix operand order repaired and checked by reading, not compiled, not run on hardware
//
#include <hip/hip_runtime.h>

// Problem constants (match reference)
#define B_   32
#define H_   32
#define KVH_ 8
#define REP_ 4            // H_/KVH_ (GQA group size)
#define D_   128
#define S_   2048
#define NWAVES 4
#define THREADS (NWAVES*32)
#define KEYS_PER_WAVE (S_/NWAVES)    // 512
#define CHUNK 32                     // keys per inner chunk (WMMA K for w*v)
#define NCHUNK (KEYS_PER_WAVE/CHUNK) // 16
#define SCALE_ 0.08838834764831845f  // 1/sqrt(128)
#define VT_STRIDE 40                 // padded row (ushorts): 80B, 16B-aligned, de-conflicted banks

typedef __attribute__((ext_vector_type(16))) __bf16 v16bf;
typedef __attribute__((ext_vector_type(8)))  float  v8f;
typedef __attribute__((ext_vector_type(4)))  float  v4f;
typedef __attribute__((ext_vector_type(4)))  unsigned int v4u;
typedef __attribute__((ext_vector_type(2)))  unsigned int v2u;

union ABReg { v16bf v; v4u q[2]; unsigned int u[8]; };

__device__ __forceinline__ unsigned short f2bf(float f) {
    unsigned int u = __float_as_uint(f);
    unsigned int r = u + 0x7FFFu + ((u >> 16) & 1u);   // round-to-nearest-even
    return (unsigned short)(r >> 16);
}

__device__ __forceinline__ void wave_fence() {
    // DS ops are in-order per wave on CDNA5; stop the compiler from reordering
    // LDS stores/loads across staging-phase boundaries.
    __builtin_amdgcn_wave_barrier();
    asm volatile("" ::: "memory");
}

__global__ __launch_bounds__(THREADS, 1)
void paged_attn_decode_kernel(const float* __restrict__ query,
                              const float* __restrict__ k_cache,
                              const float* __restrict__ v_cache,
                              const int*   __restrict__ slot_mapping,
                              const int*   __restrict__ positions,
                              float*       __restrict__ out)
{
    // Per-wave tile buffer, time-shared:
    //   K phase:  [32 keys][128 dims] bf16 (4096 ushorts, rows 256B)
    //   V phase:  [128 dims][40 keys(pad)] bf16 transposed (5120 ushorts, rows 80B)
    __shared__ __align__(16) unsigned short sh_buf[NWAVES][128 * VT_STRIDE]; // 40 KB
    __shared__ __align__(16) unsigned short sh_q[REP_][D_];                  // rope'd q (1 KB)
    __shared__ __align__(16) unsigned short sh_w[NWAVES][REP_][CHUNK];       // weights (1 KB)
    __shared__ float sh_po[NWAVES][REP_][D_];                                // partials (8 KB)
    __shared__ float sh_pm[NWAVES][REP_];
    __shared__ float sh_pl[NWAVES][REP_];
    __shared__ float sh_freq[D_/2];                                          // RoPE table

    const int wg   = blockIdx.x;
    const int b    = wg / KVH_;
    const int kvh  = wg % KVH_;
    const int tid  = threadIdx.x;
    const int wv   = tid >> 5;
    const int lane = tid & 31;

    unsigned short* kbuf = sh_buf[wv];   // K view: key*128 + d
    unsigned short* vtb  = sh_buf[wv];   // V view: dim*VT_STRIDE + key

    // ---- RoPE frequency table ----
    for (int i = tid; i < D_/2; i += THREADS)
        sh_freq[i] = __powf(10000.0f, -(float)i / 64.0f);
    __syncthreads();

    // ---- RoPE on q (position = positions[b, S-1]), store bf16 ----
    const int* slot_row = slot_mapping + (size_t)b * S_;
    const int* pos_row  = positions    + (size_t)b * S_;
    const int pos_q = pos_row[S_ - 1];
    for (int idx = tid; idx < REP_*(D_/2); idx += THREADS) {
        int hl = idx >> 6;        // local head 0..3
        int dp = idx & 63;        // dim pair 0..63
        const float* qp = query + ((size_t)(b*H_ + kvh*REP_ + hl)) * D_;
        float x1 = qp[dp], x2 = qp[dp + 64];
        float sn, cs; __sincosf((float)pos_q * sh_freq[dp], &sn, &cs);
        sh_q[hl][dp]      = f2bf(x1*cs - x2*sn);
        sh_q[hl][dp + 64] = f2bf(x2*cs + x1*sn);
    }
    __syncthreads();

    // ---- build q A-operands (reused for all key tiles) ----
    // A 16x32 bf16: lane m=l%16, hf=l/16; VGPRs 0-3 hold K=hf*8..hf*8+7,
    // VGPRs 4-7 hold K=16+hf*8.. -> two contiguous 16B runs -> 2x ds_load_b128.
    ABReg a_q[4];
    {
        const int m = lane & 15, hf = lane >> 4;
        #pragma unroll
        for (int c = 0; c < 4; ++c) {
            if (m < REP_) {
                const unsigned short* qrow = sh_q[m];
                a_q[c].q[0] = *(const v4u*)(qrow + c*32 + hf*8);
                a_q[c].q[1] = *(const v4u*)(qrow + c*32 + 16 + hf*8);
            } else {
                #pragma unroll
                for (int v = 0; v < 8; ++v) a_q[c].u[v] = 0u;
            }
        }
    }

    // ---- flash-decoding state ----
    v8f c_o[8];
    #pragma unroll
    for (int j = 0; j < 8; ++j)
        #pragma unroll
        for (int e = 0; e < 8; ++e) c_o[j][e] = 0.0f;
    float m_run[REP_], l_run[REP_];
    #pragma unroll
    for (int r = 0; r < REP_; ++r) { m_run[r] = -1e30f; l_run[r] = 0.0f; }

    const int base0 = wv * KEYS_PER_WAVE;

    for (int cc = 0; cc < NCHUNK; ++cc) {
        const int kbase = base0 + cc*CHUNK;

        // prefetch next chunk's K/V rows (one 512B row per lane)
        if (cc + 1 < NCHUNK) {
            int sp = base0 + (cc + 1)*CHUNK + lane;
            int slp = slot_row[sp];
            if (slp >= 0) {
                const float* kp = k_cache + ((size_t)slp*KVH_ + kvh) * D_;
                const float* vp = v_cache + ((size_t)slp*KVH_ + kvh) * D_;
                __builtin_prefetch(kp, 0, 1);
                __builtin_prefetch(kp + 64, 0, 1);
                __builtin_prefetch(vp, 0, 1);
                __builtin_prefetch(vp + 64, 0, 1);
            }
        }

        // ---- stage K chunk: 32 keys x 128 dims, RoPE + bf16 ----
        // float4 coalesced global loads; 2x ds_store_b64 per iter.
        #pragma unroll 2
        for (int it = 0; it < 16; ++it) {
            int idx = lane + it*32;         // 0..511
            int kl  = idx >> 4;             // key 0..31
            int qd  = (idx & 15) * 4;       // dim quad base 0..60
            int s   = kbase + kl;
            int slot = slot_row[s];
            v4f x = {0.f,0.f,0.f,0.f}, y = {0.f,0.f,0.f,0.f};
            float pos = 0.f;
            if (slot >= 0) {
                const float* p = k_cache + ((size_t)slot*KVH_ + kvh) * D_;
                x = *(const v4f*)(p + qd);
                y = *(const v4f*)(p + 64 + qd);
                pos = (float)pos_row[s];
            }
            unsigned short o1[4], o2[4];
            #pragma unroll
            for (int e = 0; e < 4; ++e) {
                float sn, cs; __sincosf(pos * sh_freq[qd + e], &sn, &cs);
                o1[e] = f2bf(x[e]*cs - y[e]*sn);
                o2[e] = f2bf(y[e]*cs + x[e]*sn);
            }
            v2u w1, w2;
            w1[0] = (unsigned int)o1[0] | ((unsigned int)o1[1] << 16);
            w1[1] = (unsigned int)o1[2] | ((unsigned int)o1[3] << 16);
            w2[0] = (unsigned int)o2[0] | ((unsigned int)o2[1] << 16);
            w2[1] = (unsigned int)o2[2] | ((unsigned int)o2[3] << 16);
            *(v2u*)(kbuf + kl*128 + qd)      = w1;
            *(v2u*)(kbuf + kl*128 + 64 + qd) = w2;
        }
        wave_fence();

        // ---- scores: two 16-key subtiles, K-dim 128 in 4 WMMAs each ----
        // B 32x16 bf16: lane n=l%16 holds K=hf*8..+7 and K=16+hf*8..+7 -> 2x ds_load_b128.
        v8f c_s[2];
        #pragma unroll
        for (int t = 0; t < 2; ++t)
            #pragma unroll
            for (int e = 0; e < 8; ++e) c_s[t][e] = 0.0f;
        {
            const int n = lane & 15, hf = lane >> 4;
            #pragma unroll
            for (int t = 0; t < 2; ++t) {
                const unsigned short* krow = kbuf + (t*16 + n)*128;
                #pragma unroll
                for (int c = 0; c < 4; ++c) {
                    ABReg bk;
                    bk.q[0] = *(const v4u*)(krow + c*32 + hf*8);
                    bk.q[1] = *(const v4u*)(krow + c*32 + 16 + hf*8);
                    c_s[t] = __builtin_amdgcn_wmma_f32_16x16x32_bf16(
                        false, a_q[c].v, false, bk.v, (short)0, c_s[t], false, false);
                }
            }
        }
        #pragma unroll
        for (int t = 0; t < 2; ++t)
            #pragma unroll
            for (int e = 0; e < 8; ++e) c_s[t][e] *= SCALE_;

        // ---- online softmax per head (rows 0..3, data in lanes 0..15) ----
        float w0[REP_], w1[REP_], alpha[REP_];
        #pragma unroll
        for (int r = 0; r < REP_; ++r) {
            float v0 = c_s[0][r], v1 = c_s[1][r];
            float mt = fmaxf(v0, v1);
            #pragma unroll
            for (int off = 1; off < 16; off <<= 1)
                mt = fmaxf(mt, __shfl_xor(mt, off, 32));
            float mn = fmaxf(m_run[r], mt);
            alpha[r] = __expf(m_run[r] - mn);
            m_run[r] = mn;
            w0[r] = __expf(v0 - mn);
            w1[r] = __expf(v1 - mn);
            float rs = w0[r] + w1[r];
            #pragma unroll
            for (int off = 1; off < 16; off <<= 1)
                rs += __shfl_xor(rs, off, 32);
            l_run[r] = l_run[r]*alpha[r] + rs;
        }
        #pragma unroll
        for (int j = 0; j < 8; ++j)
            #pragma unroll
            for (int r = 0; r < REP_; ++r) c_o[j][r] *= alpha[r];

        // weights -> LDS (C layout -> A layout goes through LDS)
        if (lane < 16) {
            #pragma unroll
            for (int r = 0; r < REP_; ++r) {
                sh_w[wv][r][lane]      = f2bf(w0[r]);
                sh_w[wv][r][16 + lane] = f2bf(w1[r]);
            }
        }
        wave_fence();

        // ---- stage V chunk TRANSPOSED: vt[dim][key], two keys packed per b32 ----
        // (reuses sh_buf; K reads already retired, DS in-order per wave)
        #pragma unroll 2
        for (int it = 0; it < 8; ++it) {
            int idx = lane + it*32;          // 0..255
            int kp2 = idx >> 4;              // key pair 0..15
            int qd  = (idx & 15) * 4;        // dim quad base
            int k0  = kp2*2;
            int sl0 = slot_row[kbase + k0];
            int sl1 = slot_row[kbase + k0 + 1];
            v4f x0 = {0.f,0.f,0.f,0.f}, y0 = {0.f,0.f,0.f,0.f};
            v4f x1 = {0.f,0.f,0.f,0.f}, y1 = {0.f,0.f,0.f,0.f};
            if (sl0 >= 0) {
                const float* p = v_cache + ((size_t)sl0*KVH_ + kvh) * D_;
                x0 = *(const v4f*)(p + qd); y0 = *(const v4f*)(p + 64 + qd);
            }
            if (sl1 >= 0) {
                const float* p = v_cache + ((size_t)sl1*KVH_ + kvh) * D_;
                x1 = *(const v4f*)(p + qd); y1 = *(const v4f*)(p + 64 + qd);
            }
            #pragma unroll
            for (int e = 0; e < 4; ++e) {
                unsigned int lo = (unsigned int)f2bf(x0[e]) | ((unsigned int)f2bf(x1[e]) << 16);
                unsigned int hi = (unsigned int)f2bf(y0[e]) | ((unsigned int)f2bf(y1[e]) << 16);
                *(unsigned int*)(vtb + (qd + e)*VT_STRIDE + k0)      = lo;
                *(unsigned int*)(vtb + (64 + qd + e)*VT_STRIDE + k0) = hi;
            }
        }
        wave_fence();

        // ---- weights A-operand: two contiguous 16B runs -> 2x ds_load_b128 ----
        ABReg aw;
        {
            const int m = lane & 15, hf = lane >> 4;
            if (m < REP_) {
                const unsigned short* wrow = sh_w[wv][m];
                aw.q[0] = *(const v4u*)(wrow + hf*8);
                aw.q[1] = *(const v4u*)(wrow + 16 + hf*8);
            } else {
                #pragma unroll
                for (int v = 0; v < 8; ++v) aw.u[v] = 0u;
            }
        }

        // ---- out += W(16x32) * V(32x16) over 8 N-tiles covering D=128 ----
        // B operand: keys hf*8..+7 and 16+hf*8..+7 of the transposed V row -> 2x ds_load_b128.
        {
            const int n = lane & 15, hf = lane >> 4;
            #pragma unroll
            for (int j = 0; j < 8; ++j) {
                const unsigned short* vrow = vtb + (j*16 + n)*VT_STRIDE;
                ABReg bv;
                bv.q[0] = *(const v4u*)(vrow + hf*8);
                bv.q[1] = *(const v4u*)(vrow + 16 + hf*8);
                c_o[j] = __builtin_amdgcn_wmma_f32_16x16x32_bf16(
                    false, aw.v, false, bv.v, (short)0, c_o[j], false, false);
            }
        }
        wave_fence();
    }

    // ---- write per-wave partials ----
    if (lane < 16) {
        #pragma unroll
        for (int r = 0; r < REP_; ++r)
            #pragma unroll
            for (int j = 0; j < 8; ++j)
                sh_po[wv][r][j*16 + lane] = c_o[j][r];
        if (lane == 0)
            #pragma unroll
            for (int r = 0; r < REP_; ++r) { sh_pm[wv][r] = m_run[r]; sh_pl[wv][r] = l_run[r]; }
    }
    __syncthreads();

    // ---- log-sum-exp merge of the 4 wave partials, normalize, store ----
    for (int idx = tid; idx < REP_*D_; idx += THREADS) {
        int r = idx >> 7;     // head 0..3
        int d = idx & 127;
        float M = -1e30f;
        #pragma unroll
        for (int w = 0; w < NWAVES; ++w) M = fmaxf(M, sh_pm[w][r]);
        float L = 0.f, O = 0.f;
        #pragma unroll
        for (int w = 0; w < NWAVES; ++w) {
            float f = __expf(sh_pm[w][r] - M);
            L += f * sh_pl[w][r];
            O += f * sh_po[w][r][d];
        }
        out[((size_t)(b*H_ + kvh*REP_ + r)) * D_ + d] = O / L;
    }
}

extern "C" void kernel_launch(void* const* d_in, const int* in_sizes, int n_in,
                              void* d_out, int out_size, void* d_ws, size_t ws_size,
                              hipStream_t stream) {
    const float* query        = (const float*)d_in[0];
    const float* k_cache      = (const float*)d_in[1];
    const float* v_cache      = (const float*)d_in[2];
    const int*   slot_mapping = (const int*)d_in[3];
    const int*   positions    = (const int*)d_in[4];
    float* out = (float*)d_out;

    dim3 grid(B_ * KVH_);     // 256 workgroups: one per (batch, kv-head)
    dim3 block(THREADS);      // 4 wave32s
    hipLaunchKernelGGL(paged_attn_decode_kernel, grid, block, 0, stream,
                       query, k_cache, v_cache, slot_mapping, positions, out);
}
